// Processer_13623636263131
// MI455X (gfx1250) — matrix operand order repaired and checked by reading, not compiled, not run on hardware
//
#include <hip/hip_runtime.h>
#include <hip/hip_bf16.h>

#define H 128
#define NNODES 50000
#define NEDGES 600000

typedef __attribute__((ext_vector_type(16))) __bf16 bf16x16;
typedef __attribute__((ext_vector_type(8)))  __bf16 bf16x8;
typedef __attribute__((ext_vector_type(8)))  float  f32x8;
typedef __attribute__((ext_vector_type(4)))  float  f32x4;

// Branch-free tanh: hardware TANH if the builtin exists, else exp2+rcp.
__device__ inline float fast_tanh(float x) {
#if __has_builtin(__builtin_amdgcn_tanhf)
    return __builtin_amdgcn_tanhf(x);
#else
    float t = __builtin_amdgcn_exp2f(x * 2.885390081777927f);
    return 1.0f - 2.0f * __builtin_amdgcn_rcpf(t + 1.0f);
#endif
}

// Non-returning fp32 atomic add -> global_atomic_add_f32 (device scope).
__device__ inline void atomic_add_f32(float* p, float v) {
    __hip_atomic_fetch_add(p, v, __ATOMIC_RELAXED, __HIP_MEMORY_SCOPE_AGENT);
}

// ---------------------------------------------------------------------------
// A-fragment from LDS bf16 tile (row-major, stride in elements).
// ISA 16-bit A 16x32 layout: lanes 0-15 row=lane hold K 0..7 & 16..23,
// lanes 16-31 row=lane-16 hold K 8..15 & 24..31.  Two 16B ds loads per lane.
// ---------------------------------------------------------------------------
__device__ inline bf16x16 load_a_frag(const __bf16* smem, int stride, int kt, int lane) {
    int r  = lane & 15;
    int hi = lane >> 4;
    int k0 = kt * 32 + hi * 8;
    const bf16x8* p0 = (const bf16x8*)(smem + r * stride + k0);
    const bf16x8* p1 = (const bf16x8*)(smem + r * stride + k0 + 16);
    bf16x8 lo = *p0;
    bf16x8 hh = *p1;
    bf16x16 a;
#pragma unroll
    for (int i = 0; i < 8; ++i) { a[i] = lo[i]; a[i + 8] = hh[i]; }
    return a;
}

// B-fragment: pre-arranged so each lane reads 16 contiguous bf16 (32 bytes).
__device__ inline bf16x16 load_b_frag(const __bf16* frag, int kt, int nt, int lane) {
    return *(const bf16x16*)(frag + ((size_t)((kt * 8 + nt) * 32 + lane) << 4));
}

__device__ inline f32x8 wmma_bf16(bf16x16 a, bf16x16 b, f32x8 c) {
    return __builtin_amdgcn_wmma_f32_16x16x32_bf16(false, a, false, b, (short)0, c,
                                                   false, false);
}

// Convert fp32 weight matrix (Ktot x 128 row-major) into bf16 B-fragments.
__global__ __launch_bounds__(256) void wconv_kernel(const float* __restrict__ w,
                                                    __bf16* __restrict__ frag,
                                                    int ktiles) {
    int t = blockIdx.x * 256 + threadIdx.x;
    if (t >= ktiles * 4096) return;
    int kt   = t >> 12;
    int r    = t & 4095;
    int nt   = r >> 9;
    int r2   = r & 511;
    int lane = r2 >> 4;
    int e    = r2 & 15;
    int col  = lane & 15;
    int k    = kt * 32 + ((lane >> 4) << 3) + (e & 7) + ((e >> 3) << 4);
    frag[t] = (__bf16)w[(size_t)k * H + nt * 16 + col];
}

__global__ __launch_bounds__(256) void zero_kernel(float* __restrict__ p, long long n4) {
    long long i = (long long)blockIdx.x * 256 + threadIdx.x;
    if (i < n4) ((f32x4*)p)[i] = (f32x4){0.f, 0.f, 0.f, 0.f};
}

// ---------------------------------------------------------------------------
// Dual-M-subtile K-loop with scheduler-enforced pipelining:
// issue order [8 DS reads][2 WMMA][4 DS reads][2 WMMA]... so every WMMA has a
// full K-step of LDS-load lookahead (forces distinct regs at RA, partial
// s_wait_dscnt instead of wait-to-zero before each WMMA).
// Masks: 0x100 = DS read, 0x008 = MFMA/WMMA.
// ---------------------------------------------------------------------------
template <int KT>
__device__ inline void mma_2m(const __bf16* sm, int stride, const bf16x16* B,
                              int lane, f32x8& acc0, f32x8& acc1) {
    bf16x16 a0 = load_a_frag(sm,               stride, 0, lane);
    bf16x16 a1 = load_a_frag(sm + 16 * stride, stride, 0, lane);
#pragma unroll
    for (int kt = 0; kt < KT; ++kt) {
        bf16x16 n0 = a0, n1 = a1;
        if (kt + 1 < KT) {
            n0 = load_a_frag(sm,               stride, kt + 1, lane);
            n1 = load_a_frag(sm + 16 * stride, stride, kt + 1, lane);
        }
        acc0 = wmma_bf16(a0, B[kt], acc0);
        acc1 = wmma_bf16(a1, B[kt], acc1);
        a0 = n0;
        a1 = n1;
    }
#if __has_builtin(__builtin_amdgcn_sched_group_barrier)
    __builtin_amdgcn_sched_group_barrier(0x100, 8, 0);   // kt0 + kt1 A-loads
#pragma unroll
    for (int i = 0; i < KT - 1; ++i) {
        __builtin_amdgcn_sched_group_barrier(0x008, 2, 0);   // 2 WMMA
        __builtin_amdgcn_sched_group_barrier(0x100, 4, 0);   // next A-loads
    }
    __builtin_amdgcn_sched_group_barrier(0x008, 2, 0);       // final 2 WMMA
#endif
}

// ---------------------------------------------------------------------------
// Edge kernel: PERSISTENT.  B-fragments for both layers live in VGPRs for the
// whole kernel; grid-stride loop over 32-edge tiles.  Each of 8 waves owns one
// 16-col output tile and two 16-row M-subtiles (2 WMMAs per held B-fragment).
// ---------------------------------------------------------------------------
#define EDGE_TILES (NEDGES / 32)   // 18750, exact

__global__ __launch_bounds__(256) void edge_kernel(
    const float* __restrict__ hidden, const int* __restrict__ ei,
    const __bf16* __restrict__ w1f, const float* __restrict__ b1,
    const __bf16* __restrict__ w2f, const float* __restrict__ b2,
    float* __restrict__ agg) {
    __shared__ __align__(16) __bf16 sA[32 * 264];   // 32 x 256 (+8 pad)
    __shared__ __align__(16) __bf16 sM[32 * 136];   // 32 x 128 (+8 pad)
    __shared__ int sDst[32];

    const int tid  = threadIdx.x;
    const int lane = tid & 31;
    const int wv   = tid >> 5;              // nt: output column tile
    const int col  = lane & 15;
    const int hi   = lane >> 4;

    // Hold all B-fragments in registers for the whole kernel.
    bf16x16 B1[8], B2[4];
#pragma unroll
    for (int kt = 0; kt < 8; ++kt) B1[kt] = load_b_frag(w1f, kt, wv, lane);
#pragma unroll
    for (int kt = 0; kt < 4; ++kt) B2[kt] = load_b_frag(w2f, kt, wv, lane);
    const float bias1 = b1[wv * 16 + col];
    const float bias2 = b2[wv * 16 + col];

    for (int tile = blockIdx.x; tile < EDGE_TILES; tile += gridDim.x) {
        const long long e0 = (long long)tile * 32;

        if (tid < 32) sDst[tid] = ei[NEDGES + e0 + tid];
        // Gather + fp32->bf16: thread covers 32 cols of one row.
        {
            int m = tid >> 3;               // row 0..31
            int q = tid & 7;                // 32-col chunk 0..7
            long long e = e0 + m;
            int node = (q < 4) ? ei[e] : ei[NEDGES + e];
            const float* srow = hidden + (long long)node * H + (q & 3) * 32;
            __bf16* drow = sA + m * 264 + q * 32;
#pragma unroll
            for (int i = 0; i < 32; i += 4) {
                f32x4 v = *(const f32x4*)(srow + i);
                drow[i + 0] = (__bf16)v.x;
                drow[i + 1] = (__bf16)v.y;
                drow[i + 2] = (__bf16)v.z;
                drow[i + 3] = (__bf16)v.w;
            }
        }
        __syncthreads();

        // Layer 1: K = 256 (pipelined).
        f32x8 acc0 = {}, acc1 = {};
        mma_2m<8>(sA, 264, B1, lane, acc0, acc1);
#pragma unroll
        for (int r = 0; r < 8; ++r) {
            int row = r + hi * 8;
            sM[row * 136 + wv * 16 + col]        = (__bf16)fast_tanh(acc0[r] + bias1);
            sM[(16 + row) * 136 + wv * 16 + col] = (__bf16)fast_tanh(acc1[r] + bias1);
        }
        __syncthreads();

        // Layer 2: K = 128 (pipelined).
        f32x8 c0 = {}, c1 = {};
        mma_2m<4>(sM, 136, B2, lane, c0, c1);
#pragma unroll
        for (int r = 0; r < 8; ++r) {
            int row = r + hi * 8;
            atomic_add_f32(&agg[(long long)sDst[row] * H + wv * 16 + col],
                           c0[r] + bias2);
            atomic_add_f32(&agg[(long long)sDst[16 + row] * H + wv * 16 + col],
                           c1[r] + bias2);
        }
        __syncthreads();   // protect sA/sM/sDst before next iteration
    }
}

// ---------------------------------------------------------------------------
// Node kernel: 32 nodes per block, concat(hidden, agg) -> 3-layer MLP.
// Per-layer B-fragments hoisted into registers ahead of each pipelined K-loop.
// ---------------------------------------------------------------------------
__global__ __launch_bounds__(256) void node_kernel(
    const float* __restrict__ hidden, const float* __restrict__ agg,
    const __bf16* __restrict__ w1f, const float* __restrict__ b1,
    const __bf16* __restrict__ w2f, const float* __restrict__ b2,
    const __bf16* __restrict__ w3f, const float* __restrict__ b3,
    float* __restrict__ out) {
    __shared__ __align__(16) __bf16 sA[32 * 264];
    __shared__ __align__(16) __bf16 sM1[32 * 136];
    __shared__ __align__(16) __bf16 sM2[32 * 136];

    const int tid  = threadIdx.x;
    const int lane = tid & 31;
    const int wv   = tid >> 5;
    const int col  = lane & 15;
    const int hi   = lane >> 4;
    const long long n0 = (long long)blockIdx.x * 32;

    {
        int m = tid >> 3;
        int q = tid & 7;
        long long node = n0 + m;
        if (node >= NNODES) node = NNODES - 1;
        const float* base = (q < 4) ? hidden : agg;
        const float* srow = base + node * H + (q & 3) * 32;
        __bf16* drow = sA + m * 264 + q * 32;
#pragma unroll
        for (int i = 0; i < 32; i += 4) {
            f32x4 v = *(const f32x4*)(srow + i);
            drow[i + 0] = (__bf16)v.x;
            drow[i + 1] = (__bf16)v.y;
            drow[i + 2] = (__bf16)v.z;
            drow[i + 3] = (__bf16)v.w;
        }
    }
    __syncthreads();

    // Layer 1: K = 256
    {
        bf16x16 B1[8];
#pragma unroll
        for (int kt = 0; kt < 8; ++kt) B1[kt] = load_b_frag(w1f, kt, wv, lane);
        f32x8 acc0 = {}, acc1 = {};
        mma_2m<8>(sA, 264, B1, lane, acc0, acc1);
        float bias = b1[wv * 16 + col];
#pragma unroll
        for (int r = 0; r < 8; ++r) {
            int row = r + hi * 8;
            sM1[row * 136 + wv * 16 + col]        = (__bf16)fast_tanh(acc0[r] + bias);
            sM1[(16 + row) * 136 + wv * 16 + col] = (__bf16)fast_tanh(acc1[r] + bias);
        }
    }
    __syncthreads();

    // Layer 2: K = 128
    {
        bf16x16 B2[4];
#pragma unroll
        for (int kt = 0; kt < 4; ++kt) B2[kt] = load_b_frag(w2f, kt, wv, lane);
        f32x8 c0 = {}, c1 = {};
        mma_2m<4>(sM1, 136, B2, lane, c0, c1);
        float bias = b2[wv * 16 + col];
#pragma unroll
        for (int r = 0; r < 8; ++r) {
            int row = r + hi * 8;
            sM2[row * 136 + wv * 16 + col]        = (__bf16)fast_tanh(c0[r] + bias);
            sM2[(16 + row) * 136 + wv * 16 + col] = (__bf16)fast_tanh(c1[r] + bias);
        }
    }
    __syncthreads();

    // Layer 3: K = 128, no tanh, fp32 output.
    {
        bf16x16 B3[4];
#pragma unroll
        for (int kt = 0; kt < 4; ++kt) B3[kt] = load_b_frag(w3f, kt, wv, lane);
        f32x8 d0 = {}, d1 = {};
        mma_2m<4>(sM2, 136, B3, lane, d0, d1);
        float bias = b3[wv * 16 + col];
#pragma unroll
        for (int r = 0; r < 8; ++r) {
            long long row0 = n0 + r + hi * 8;
            long long row1 = row0 + 16;
            if (row0 < NNODES) out[row0 * H + wv * 16 + col] = d0[r] + bias;
            if (row1 < NNODES) out[row1 * H + wv * 16 + col] = d1[r] + bias;
        }
    }
}

extern "C" void kernel_launch(void* const* d_in, const int* in_sizes, int n_in,
                              void* d_out, int out_size, void* d_ws, size_t ws_size,
                              hipStream_t stream) {
    const float* hidden = (const float*)d_in[0];
    const int*   ei     = (const int*)d_in[1];
    const float* ew1 = (const float*)d_in[2];
    const float* eb1 = (const float*)d_in[3];
    const float* ew2 = (const float*)d_in[4];
    const float* eb2 = (const float*)d_in[5];
    const float* uw1 = (const float*)d_in[6];
    const float* ub1 = (const float*)d_in[7];
    const float* uw2 = (const float*)d_in[8];
    const float* ub2 = (const float*)d_in[9];
    const float* uw3 = (const float*)d_in[10];
    const float* ub3 = (const float*)d_in[11];
    float* out = (float*)d_out;

    // Workspace: agg[N*H] f32, then bf16 weight fragments.
    float*  agg  = (float*)d_ws;
    __bf16* wbuf = (__bf16*)((char*)d_ws + (size_t)NNODES * H * sizeof(float));
    __bf16* ew1f = wbuf;                    // 256*128
    __bf16* ew2f = ew1f + 256 * 128;        // 128*128
    __bf16* uw1f = ew2f + 128 * 128;        // 256*128
    __bf16* uw2f = uw1f + 256 * 128;        // 128*128
    __bf16* uw3f = uw2f + 128 * 128;        // 128*128

    // Re-zero agg every launch (graph replay safe).
    long long n4 = (long long)NNODES * H / 4;
    zero_kernel<<<(int)((n4 + 255) / 256), 256, 0, stream>>>(agg, n4);

    // Convert weights to bf16 fragments (ktiles = Ktot/32).
    wconv_kernel<<<8 * 16, 256, 0, stream>>>(ew1, ew1f, 8);
    wconv_kernel<<<4 * 16, 256, 0, stream>>>(ew2, ew2f, 4);
    wconv_kernel<<<8 * 16, 256, 0, stream>>>(uw1, uw1f, 8);
    wconv_kernel<<<4 * 16, 256, 0, stream>>>(uw2, uw2f, 4);
    wconv_kernel<<<4 * 16, 256, 0, stream>>>(uw3, uw3f, 4);

    // Persistent edge kernel: 1875 blocks x 10 tiles of 32 edges.
    edge_kernel<<<1875, 256, 0, stream>>>(hidden, ei, ew1f, eb1, ew2f, eb2, agg);
    node_kernel<<<(NNODES + 31) / 32, 256, 0, stream>>>(hidden, agg, uw1f, ub1,
                                                        uw2f, ub2, uw3f, ub3, out);
}
